// PerceiverResampler_66305705116357
// MI455X (gfx1250) — compile-verified, matrix operand order checked by reading, not run on hardware
//
#include <hip/hip_runtime.h>

// ---------------------------------------------------------------------------
// Perceiver Resampler forward for MI455X (gfx1250).
// bf16 WMMA for all matmuls, async global->LDS (ASYNCcnt) tile staging.
// B=32, S=1024, D_IN=1280, D=1024, NH=8, DH=128, NLAT=64, NL=2, DFF=4096
// ---------------------------------------------------------------------------

typedef __attribute__((ext_vector_type(8)))  __bf16 v8bf;
typedef __attribute__((ext_vector_type(16))) __bf16 v16bf;
typedef __attribute__((ext_vector_type(8)))  float  v8f;

#define WMMA_BF16(a, b, c) \
  __builtin_amdgcn_wmma_f32_16x16x32_bf16(false, (a), false, (b), (short)0, (c), false, false)

// One 16-byte-per-lane async DMA chunk: global -> LDS, tracked by ASYNCcnt.
static __device__ __forceinline__ void async_ld_b128(unsigned lds_off,
                                                     const void* g) {
  asm volatile("global_load_async_to_lds_b128 %0, %1, off"
               :
               : "v"(lds_off), "v"((unsigned long long)(uintptr_t)g)
               : "memory");
}
static __device__ __forceinline__ unsigned lds_addr(const void* p) {
  return (unsigned)(uintptr_t)p;  // low 32 bits of shared-aperture pointer
}

// Load one 16x32 A/B fragment for v_wmma_f32_16x16x32_bf16.
// Lane (l&15) = row/col index; element e maps to
// K = (e/8)*16 + (l>>4)*8 + (e%8)  -> two contiguous 8-elem (16B) runs.
// p0 must point at row_base + half*8.
static __device__ __forceinline__ v16bf load_frag(const __bf16* p0) {
  v8bf lo = *(const v8bf*)(p0);
  v8bf hi = *(const v8bf*)(p0 + 16);
  return __builtin_shufflevector(lo, hi, 0, 1, 2, 3, 4, 5, 6, 7,
                                 8, 9, 10, 11, 12, 13, 14, 15);
}

// ---------------------------------------------------------------------------
// bf16 GEMM:  C[M,N] (f32 accum) = A[M,K] * BT[N,K]^T   (BT pre-transposed)
// Block tile 128x128, 256 threads = 8 waves (4M x 2N), wave tile 32x64.
// Double-buffered LDS, staged entirely with global_load_async_to_lds_b128.
// EPI: 0 = store f32 (+bias)       1 = store bf16 (+bias)
//      2 = gelu(bias+acc) -> bf16  3 = Cf += (bias+acc)  (residual)
// ---------------------------------------------------------------------------
template <int EPI>
__global__ __launch_bounds__(256) void gemm_bf16(
    const __bf16* __restrict__ A, const __bf16* __restrict__ BT,
    const float* __restrict__ bias, float* __restrict__ Cf,
    __bf16* __restrict__ Cb, int M, int N, int K) {
  __shared__ __align__(16) __bf16 lds_a[2][128 * 32];  // [buf][row][k]
  __shared__ __align__(16) __bf16 lds_b[2][128 * 32];  // [buf][col][k]

  const int t = threadIdx.x;
  const int lane = t & 31, half = lane >> 4, ln = lane & 15;
  const int w = t >> 5, wm = w & 3, wn = w >> 2;
  const int m0 = blockIdx.y * 128, n0 = blockIdx.x * 128;

  const v8f vz = {0.f, 0.f, 0.f, 0.f, 0.f, 0.f, 0.f, 0.f};
  v8f acc[2][4];
#pragma unroll
  for (int i = 0; i < 2; ++i)
#pragma unroll
    for (int j = 0; j < 4; ++j) acc[i][j] = vz;

  const int sr = t >> 1;          // staging row/col 0..127
  const int sk = (t & 1) * 16;    // staging k-base (elements)

  // Stage one 128x32 A tile + one 128x32 B tile: 4 async b128 per thread.
  auto stage = [&](int k0, int buf) {
    const __bf16* ga = A + (size_t)(m0 + sr) * K + k0 + sk;
    const __bf16* gb = BT + (size_t)(n0 + sr) * K + k0 + sk;
    unsigned la = lds_addr(&lds_a[buf][sr * 32 + sk]);
    unsigned lb = lds_addr(&lds_b[buf][sr * 32 + sk]);
    async_ld_b128(la, ga);
    async_ld_b128(la + 16, ga + 8);
    async_ld_b128(lb, gb);
    async_ld_b128(lb + 16, gb + 8);
  };

  stage(0, 0);
  const int nk = K >> 5;
  for (int ki = 0; ki < nk; ++ki) {
    if (ki + 1 < nk) {
      stage((ki + 1) << 5, (ki + 1) & 1);
      // loads complete in order: <=4 outstanding means tile ki has landed
      asm volatile("s_wait_asynccnt 0x4" ::: "memory");
    } else {
      asm volatile("s_wait_asynccnt 0x0" ::: "memory");
    }
    __syncthreads();
    const int buf = ki & 1;
    v16bf bf[4];
#pragma unroll
    for (int j = 0; j < 4; ++j)
      bf[j] = load_frag(&lds_b[buf][(wn * 64 + j * 16 + ln) * 32 + half * 8]);
#pragma unroll
    for (int i = 0; i < 2; ++i) {
      v16bf af = load_frag(&lds_a[buf][(wm * 32 + i * 16 + ln) * 32 + half * 8]);
#pragma unroll
      for (int j = 0; j < 4; ++j) acc[i][j] = WMMA_BF16(af, bf[j], acc[i][j]);
    }
    __syncthreads();
  }

  // epilogue: D layout -> lane holds (M = r + 8*half, N = ln) per VGPR r
#pragma unroll
  for (int i = 0; i < 2; ++i)
#pragma unroll
    for (int j = 0; j < 4; ++j)
#pragma unroll
      for (int r = 0; r < 8; ++r) {
        int gm = m0 + wm * 32 + i * 16 + r + 8 * half;
        int gn = n0 + wn * 64 + j * 16 + ln;
        float v = acc[i][j][r];
        if (bias) v += bias[gn];
        size_t idx = (size_t)gm * N + gn;
        if (EPI == 0) {
          Cf[idx] = v;
        } else if (EPI == 1) {
          Cb[idx] = (__bf16)v;
        } else if (EPI == 2) {
          float g = 0.5f * v * (1.0f + erff(v * 0.70710678118654752f));
          Cb[idx] = (__bf16)g;
        } else {
          Cf[idx] += v;
        }
      }
}

// ---------------------------------------------------------------------------
// Fused attention per (batch, head). grid.x = 32*8, block = 128 (4 waves).
// Wave w owns latent rows [16w, 16w+16). Streams S in chunks of 32.
// Clip(-11,11) bounds scores -> plain exp/sum softmax (no running max).
// q:  [B*64][1024] bf16      kv: [B*S][2048] bf16 (k = cols 0..1023, roped)
// vT: [B][H][128][1024] bf16 ob: [B*64][1024] bf16
// ---------------------------------------------------------------------------
__global__ __launch_bounds__(128) void attn_kernel(
    const __bf16* __restrict__ q, const __bf16* __restrict__ kv,
    const __bf16* __restrict__ vT, __bf16* __restrict__ ob) {
  __shared__ __align__(16) __bf16 plds[4][16 * 32];
  const int b = blockIdx.x >> 3, hh = blockIdx.x & 7;
  const int wv = threadIdx.x >> 5, lane = threadIdx.x & 31;
  const int half = lane >> 4, ln = lane & 15;
  const int m0 = wv * 16;
  __bf16* pl = plds[wv];

  v16bf qf[4];
#pragma unroll
  for (int kc = 0; kc < 4; ++kc)
    qf[kc] = load_frag(q + ((size_t)(b * 64 + m0 + ln)) * 1024 + hh * 128 +
                       kc * 32 + half * 8);

  const v8f vz = {0.f, 0.f, 0.f, 0.f, 0.f, 0.f, 0.f, 0.f};
  v8f oacc[8];
#pragma unroll
  for (int dt = 0; dt < 8; ++dt) oacc[dt] = vz;
  float dsum[8] = {0.f, 0.f, 0.f, 0.f, 0.f, 0.f, 0.f, 0.f};

  for (int s0 = 0; s0 < 1024; s0 += 32) {
#pragma unroll
    for (int st = 0; st < 2; ++st) {
      v8f sacc = vz;
      const int scol = s0 + st * 16;
#pragma unroll
      for (int kc = 0; kc < 4; ++kc) {
        const __bf16* kp = kv + ((size_t)(b * 1024 + scol + ln)) * 2048 +
                           hh * 128 + kc * 32 + half * 8;
        sacc = WMMA_BF16(qf[kc], load_frag(kp), sacc);
      }
#pragma unroll
      for (int r = 0; r < 8; ++r) {
        float v = sacc[r] * 0.08838834764831845f;  // 1/sqrt(128)
        v = fminf(fmaxf(v, -11.f), 11.f);
        float e = __expf(v);
        dsum[r] += e;
        pl[(r + 8 * half) * 32 + st * 16 + ln] = (__bf16)e;
      }
    }
    v16bf pf = load_frag(pl + ln * 32 + half * 8);
#pragma unroll
    for (int dt = 0; dt < 8; ++dt) {
      const __bf16* vp = vT +
                         (((size_t)(b * 8 + hh) * 128) + dt * 16 + ln) * 1024 +
                         s0 + half * 8;
      oacc[dt] = WMMA_BF16(pf, load_frag(vp), oacc[dt]);
    }
  }
#pragma unroll
  for (int r = 0; r < 8; ++r) {
    float dv = dsum[r];
    dv += __shfl_xor(dv, 1, 32);
    dv += __shfl_xor(dv, 2, 32);
    dv += __shfl_xor(dv, 4, 32);
    dv += __shfl_xor(dv, 8, 32);
    dsum[r] = dv;
  }
#pragma unroll
  for (int dt = 0; dt < 8; ++dt)
#pragma unroll
    for (int r = 0; r < 8; ++r) {
      float v = oacc[dt][r] / dsum[r];
      ob[((size_t)(b * 64 + m0 + r + 8 * half)) * 1024 + hh * 128 + dt * 16 +
         ln] = (__bf16)v;
    }
}

// ---------------------------------------------------------------------------
// LayerNorm (block per row, D=1024), optional bf16 output.
// ---------------------------------------------------------------------------
template <bool BF16OUT>
__global__ __launch_bounds__(256) void layernorm_k(
    const float* __restrict__ x, const float* __restrict__ g,
    const float* __restrict__ bta, void* __restrict__ outp, int D) {
  const size_t row = blockIdx.x;
  const float* rx = x + row * D;
  float s = 0.f, s2 = 0.f;
  for (int i = threadIdx.x; i < D; i += blockDim.x) {
    float v = rx[i];
    s += v;
    s2 += v * v;
  }
#pragma unroll
  for (int off = 16; off > 0; off >>= 1) {
    s += __shfl_down(s, off, 32);
    s2 += __shfl_down(s2, off, 32);
  }
  __shared__ float rs[8], rs2[8];
  int w = threadIdx.x >> 5;
  if ((threadIdx.x & 31) == 0) {
    rs[w] = s;
    rs2[w] = s2;
  }
  __syncthreads();
  if (threadIdx.x == 0) {
    float ts = 0.f, ts2 = 0.f;
    for (int i = 0; i < (int)(blockDim.x >> 5); ++i) {
      ts += rs[i];
      ts2 += rs2[i];
    }
    rs[0] = ts;
    rs2[0] = ts2;
  }
  __syncthreads();
  float mean = rs[0] / (float)D;
  float var = rs2[0] / (float)D - mean * mean;
  float inv = rsqrtf(var + 1e-5f);
  for (int i = threadIdx.x; i < D; i += blockDim.x) {
    float v = (rx[i] - mean) * inv * g[i] + bta[i];
    if (BF16OUT)
      ((__bf16*)outp)[row * D + i] = (__bf16)v;
    else
      ((float*)outp)[row * D + i] = v;
  }
}

// ---------------------------------------------------------------------------
// Utility kernels
// ---------------------------------------------------------------------------
__global__ void cvt_f32_bf16(const float* __restrict__ s,
                             __bf16* __restrict__ d, long n) {
  long i = (long)blockIdx.x * blockDim.x + threadIdx.x;
  long stride = (long)gridDim.x * blockDim.x;
  for (; i < n; i += stride) d[i] = (__bf16)s[i];
}

// w[K][N] f32  ->  wT[N][K] bf16   (32x32 LDS tiles, coalesced both sides)
__global__ __launch_bounds__(256) void cvt_transpose(
    const float* __restrict__ w, __bf16* __restrict__ wT, int K, int N) {
  __shared__ float tile[32][33];
  const int k0 = blockIdx.y * 32, n0 = blockIdx.x * 32;
  const int tx = threadIdx.x & 31, ty = threadIdx.x >> 5;
#pragma unroll
  for (int r = ty; r < 32; r += 8)
    tile[r][tx] = w[(size_t)(k0 + r) * N + n0 + tx];
  __syncthreads();
#pragma unroll
  for (int r = ty; r < 32; r += 8)
    wT[(size_t)(n0 + r) * K + k0 + tx] = (__bf16)tile[tx][r];
}

__global__ void bcast_lat(const float* __restrict__ latents,
                          float* __restrict__ lat) {
  int i = blockIdx.x * blockDim.x + threadIdx.x;  // over 64*1024
  float v = latents[i];
#pragma unroll
  for (int b = 0; b < 32; ++b) lat[(size_t)b * 65536 + i] = v;
}

// cos/sin tables for 2D RoPE over DH/2 = 64 dims, S=1024 (32x32 grid)
__global__ void rope_tables(float* __restrict__ cosb, float* __restrict__ sinb) {
  int i = blockIdx.x * blockDim.x + threadIdx.x;  // over 1024*64
  int s = i >> 6, d = i & 63;
  float inv = powf(10000.f, -(float)(d & 15) / 16.f);
  float pos = (d < 32) ? (float)(s >> 5) : (float)(s & 31);  // y then x
  float e = pos * inv;
  cosb[i] = cosf(e);
  sinb[i] = sinf(e);
}

// apply RoPE to k half of kv (in place); one thread per (b,s,h,d<64)
__global__ void rope_apply(__bf16* __restrict__ kvb,
                           const float* __restrict__ cosb,
                           const float* __restrict__ sinb) {
  int i = blockIdx.x * blockDim.x + threadIdx.x;  // over 32*1024*8*64
  int d = i & 63;
  int h = (i >> 6) & 7;
  int s = (i >> 9) & 1023;
  int b = i >> 19;
  size_t base = ((size_t)(b * 1024 + s)) * 2048 + h * 128;
  float k1 = (float)kvb[base + d];
  float k2 = (float)kvb[base + 64 + d];
  float c = cosb[s * 64 + d], sn = sinb[s * 64 + d];
  kvb[base + d] = (__bf16)(k1 * c - k2 * sn);
  kvb[base + 64 + d] = (__bf16)(k2 * c + k1 * sn);
}

// v half of kv -> vT[b][h][d][s]  (coalesced writes)
__global__ void build_vT(const __bf16* __restrict__ kvb,
                         __bf16* __restrict__ vT) {
  size_t i = (size_t)blockIdx.x * blockDim.x + threadIdx.x;  // over 32M
  int s = (int)(i & 1023);
  int dd = (int)((i >> 10) & 127);
  int h = (int)((i >> 17) & 7);
  int b = (int)(i >> 20);
  vT[i] = kvb[((size_t)(b * 1024 + s)) * 2048 + 1024 + h * 128 + dd];
}

// ---------------------------------------------------------------------------
// Host orchestration
// ---------------------------------------------------------------------------
extern "C" void kernel_launch(void* const* d_in, const int* in_sizes, int n_in,
                              void* d_out, int out_size, void* d_ws,
                              size_t ws_size, hipStream_t stream) {
  (void)in_sizes;
  (void)n_in;
  (void)out_size;
  (void)ws_size;

  const float* x = (const float*)d_in[0];
  const float* in_w = (const float*)d_in[1];
  const float* in_b = (const float*)d_in[2];
  const float* latents = (const float*)d_in[3];
  const float* out_g = (const float*)d_in[4 + 2 * 14];
  const float* out_b = (const float*)d_in[5 + 2 * 14];
  float* out = (float*)d_out;

  char* base = (char*)d_ws;
  size_t off = 0;
  auto alloc = [&](size_t bytes) -> void* {
    off = (off + 255) & ~(size_t)255;
    void* p = base + off;
    off += bytes;
    return p;
  };
  float* ctx = (float*)alloc((size_t)32768 * 1024 * 4);    // ctx f32
  __bf16* kvb = (__bf16*)alloc((size_t)32768 * 2048 * 2);  // kv bf16
  __bf16* ctxn = (__bf16*)alloc((size_t)32768 * 1024 * 2); // LN(ctx) bf16
  __bf16* vT = (__bf16*)alloc((size_t)32 * 8 * 128 * 1024 * 2);
  __bf16* big = (__bf16*)alloc((size_t)32768 * 1280 * 2);  // x_bf / h_bf
  __bf16* wbuf = (__bf16*)alloc((size_t)4096 * 1024 * 2);  // wT bf16
  float* lat = (float*)alloc((size_t)2048 * 1024 * 4);
  __bf16* latn = (__bf16*)alloc((size_t)2048 * 1024 * 2);
  __bf16* qb = (__bf16*)alloc((size_t)2048 * 1024 * 2);
  __bf16* attnb = (__bf16*)alloc((size_t)2048 * 1024 * 2);
  float* cosb = (float*)alloc((size_t)1024 * 64 * 4);
  float* sinb = (float*)alloc((size_t)1024 * 64 * 4);

  // ctx = x @ in_w + in_b
  cvt_f32_bf16<<<4096, 256, 0, stream>>>(x, big, (long)32768 * 1280);
  cvt_transpose<<<dim3(32, 40), 256, 0, stream>>>(in_w, wbuf, 1280, 1024);
  gemm_bf16<0><<<dim3(8, 256), 256, 0, stream>>>(big, wbuf, in_b, ctx, nullptr,
                                                 32768, 1024, 1280);
  rope_tables<<<(1024 * 64) / 256, 256, 0, stream>>>(cosb, sinb);
  bcast_lat<<<(64 * 1024) / 256, 256, 0, stream>>>(latents, lat);

  for (int l = 0; l < 2; ++l) {
    int bi = 4 + l * 14;
    const float* nl_g = (const float*)d_in[bi + 0];
    const float* nl_b = (const float*)d_in[bi + 1];
    const float* nc_g = (const float*)d_in[bi + 2];
    const float* nc_b = (const float*)d_in[bi + 3];
    const float* wq = (const float*)d_in[bi + 4];
    const float* wkv = (const float*)d_in[bi + 5];
    const float* wo = (const float*)d_in[bi + 6];
    const float* bo = (const float*)d_in[bi + 7];
    const float* fg = (const float*)d_in[bi + 8];
    const float* fb = (const float*)d_in[bi + 9];
    const float* w1 = (const float*)d_in[bi + 10];
    const float* b1 = (const float*)d_in[bi + 11];
    const float* w2 = (const float*)d_in[bi + 12];
    const float* b2 = (const float*)d_in[bi + 13];

    // kv = LN(ctx) @ wkv ; rope(k) ; vT
    layernorm_k<true><<<32768, 256, 0, stream>>>(ctx, nc_g, nc_b, ctxn, 1024);
    cvt_transpose<<<dim3(64, 32), 256, 0, stream>>>(wkv, wbuf, 1024, 2048);
    gemm_bf16<1><<<dim3(16, 256), 256, 0, stream>>>(ctxn, wbuf, nullptr,
                                                    nullptr, kvb, 32768, 2048,
                                                    1024);
    rope_apply<<<(32 * 1024 * 8 * 64) / 256, 256, 0, stream>>>(kvb, cosb, sinb);
    build_vT<<<131072, 256, 0, stream>>>(kvb, vT);

    // q = LN(lat) @ wq
    layernorm_k<true><<<2048, 256, 0, stream>>>(lat, nl_g, nl_b, latn, 1024);
    cvt_transpose<<<dim3(32, 32), 256, 0, stream>>>(wq, wbuf, 1024, 1024);
    gemm_bf16<1><<<dim3(8, 16), 256, 0, stream>>>(latn, wbuf, nullptr, nullptr,
                                                  qb, 2048, 1024, 1024);

    // attention + output projection (residual into lat)
    attn_kernel<<<256, 128, 0, stream>>>(qb, kvb, vT, attnb);
    cvt_transpose<<<dim3(32, 32), 256, 0, stream>>>(wo, wbuf, 1024, 1024);
    gemm_bf16<3><<<dim3(8, 16), 256, 0, stream>>>(attnb, wbuf, bo, lat, nullptr,
                                                  2048, 1024, 1024);

    // FFN (residual into lat)
    layernorm_k<true><<<2048, 256, 0, stream>>>(lat, fg, fb, latn, 1024);
    cvt_transpose<<<dim3(128, 32), 256, 0, stream>>>(w1, wbuf, 1024, 4096);
    gemm_bf16<2><<<dim3(32, 16), 256, 0, stream>>>(latn, wbuf, b1, nullptr, big,
                                                   2048, 4096, 1024);
    cvt_transpose<<<dim3(32, 128), 256, 0, stream>>>(w2, wbuf, 4096, 1024);
    gemm_bf16<3><<<dim3(8, 16), 256, 0, stream>>>(big, wbuf, b2, lat, nullptr,
                                                  2048, 1024, 4096);
  }

  layernorm_k<false><<<2048, 256, 0, stream>>>(lat, out_g, out_b, out, 1024);
}
